// zBlock_19198503813393
// MI455X (gfx1250) — compile-verified, matrix operand order checked by reading, not run on hardware
//
#include <hip/hip_runtime.h>
#include <hip/hip_bf16.h>
#include <stdint.h>

// ---------------------------------------------------------------------------
// AF2 pair-stack block on MI455X (gfx1250, wave32, WMMA bf16 K=32).
// All matmuls -> v_wmma_f32_16x16x32_bf16 with 2x2 register blocking
// (4 WMMAs per 4 fragment loads); attention is flash-style so the 268MB
// logits tensor is never materialized; everything stays L2-resident.
// ---------------------------------------------------------------------------

typedef __attribute__((ext_vector_type(16))) __bf16 v16bf;
typedef __attribute__((ext_vector_type(8)))  float  v8f;
typedef __attribute__((ext_vector_type(4)))  __bf16 bf16x4;

#define LSEQ 256
#define ZC   128
#define LL   65536   // LSEQ*LSEQ

union FragU { v16bf v; uint32_t u[8]; };

__device__ __forceinline__ v8f wmma_bf16(const FragU& a, const FragU& b, v8f c) {
    return __builtin_amdgcn_wmma_f32_16x16x32_bf16(false, a.v, false, b.v,
                                                   (short)0, c, false, false);
}

// Load a 16x32 A/B fragment (ISA 7.12.2 16-bit layout) from row-major [row][K]
// bf16 storage. Caller passes this lane's row (base + lane%16) and hi=lane>=16.
// VGPR v<4 holds K pair v*2 (+hi*8); v>=4 holds K pair 16+(v-4)*2 (+hi*8):
// all pairs contiguous -> 8 packed dword loads (compiler merges to b128s).
__device__ __forceinline__ FragU load_frag(const __bf16* base, int row,
                                           int rowStride, int k0, int hi) {
    FragU f;
    const uint32_t* p = (const uint32_t*)(base + (size_t)row * rowStride + k0 + hi * 8);
#pragma unroll
    for (int v = 0; v < 4; ++v) f.u[v] = p[v];
#pragma unroll
    for (int v = 4; v < 8; ++v) f.u[v] = p[v + 4];
    return f;
}

__device__ __forceinline__ float sigmoidf_(float x) { return 1.0f / (1.0f + __expf(-x)); }

// ---------------------------------------------------------------------------
// LayerNorm over 128 channels, one wave per row. chanStride==1: contiguous
// [r][c] fp32 input; else channel-major input x[c*chanStride + r]. bf16 out.
// ---------------------------------------------------------------------------
__global__ __launch_bounds__(256) void ln_kernel(const float* __restrict__ x,
        const float* __restrict__ gam, const float* __restrict__ bet,
        __bf16* __restrict__ out, int chanStride)
{
    int lane = threadIdx.x & 31;
    int r = blockIdx.x * 8 + (threadIdx.x >> 5);
    float v[4];
    if (chanStride == 1) {
        float4 f = ((const float4*)(x + (size_t)r * ZC))[lane];
        v[0] = f.x; v[1] = f.y; v[2] = f.z; v[3] = f.w;
    } else {
#pragma unroll
        for (int q = 0; q < 4; ++q)
            v[q] = x[(size_t)(lane * 4 + q) * (size_t)chanStride + r];
    }
    float s = v[0] + v[1] + v[2] + v[3];
#pragma unroll
    for (int off = 16; off >= 1; off >>= 1) s += __shfl_xor(s, off, 32);
    float mu = s * (1.0f / 128.0f);
    float d = 0.0f;
#pragma unroll
    for (int q = 0; q < 4; ++q) { float t = v[q] - mu; d += t * t; }
#pragma unroll
    for (int off = 16; off >= 1; off >>= 1) d += __shfl_xor(d, off, 32);
    float inv = rsqrtf(d * (1.0f / 128.0f) + 1e-5f);
    bf16x4 o;
#pragma unroll
    for (int q = 0; q < 4; ++q) {
        int c = lane * 4 + q;
        o[q] = (__bf16)((v[q] - mu) * inv * gam[c] + bet[c]);
    }
    *(bf16x4*)(out + (size_t)r * ZC + lane * 4) = o;
}

// Convert fp32 W[K][N] -> bf16 Wt[N][K] (K contiguous for B-fragment loads).
__global__ __launch_bounds__(256) void wconv_kernel(const float* __restrict__ w,
        __bf16* __restrict__ wt, int K, int N)
{
    int t = blockIdx.x * 256 + threadIdx.x;
    if (t >= K * N) return;
    int k = t / N, n = t - k * N;
    wt[(size_t)n * K + k] = (__bf16)w[t];
}

// ---------------------------------------------------------------------------
// Gated projection: out = sigmoid(A@Wg+bg) * (A@Wp+bp). 2(M)x1(N) blocking:
// per k-step 4 fragment loads (2 A rows + Wg + Wp) feed 4 WMMAs.
// Scatter to channel-major for the triangle einsum:
//   mode 1 (outgoing): out[c][r]            (r = i*256+k -> [c][i][k])
//   mode 2 (incoming): out[c][r%256][r/256] (r = k*256+i -> [c][i][k])
// M=65536, N=K=128. 32x16 tile per wave, 8 waves/block -> 2048 blocks.
// ---------------------------------------------------------------------------
__global__ __launch_bounds__(256) void gemm_gated_kernel(
        const __bf16* __restrict__ A,
        const __bf16* __restrict__ WgT, const float* __restrict__ bg,
        const __bf16* __restrict__ WpT, const float* __restrict__ bp,
        __bf16* __restrict__ outp, int scatterMode)
{
    const int N = 128, K = 128;
    int lane = threadIdx.x & 31, hi = lane >> 4, ln16 = lane & 15;
    int tile = blockIdx.x * 8 + (threadIdx.x >> 5);
    int tn = N >> 4;
    int m0 = (tile / tn) << 5, n0 = (tile % tn) << 4;
    v8f aG0 = {}, aG1 = {}, aP0 = {}, aP1 = {};
    for (int k0 = 0; k0 < K; k0 += 32) {
        FragU fa0 = load_frag(A, m0 + ln16,      K, k0, hi);
        FragU fa1 = load_frag(A, m0 + 16 + ln16, K, k0, hi);
        FragU fg  = load_frag(WgT, n0 + ln16, K, k0, hi);
        FragU fp  = load_frag(WpT, n0 + ln16, K, k0, hi);
        aG0 = wmma_bf16(fa0, fg, aG0);
        aP0 = wmma_bf16(fa0, fp, aP0);
        aG1 = wmma_bf16(fa1, fg, aG1);
        aP1 = wmma_bf16(fa1, fp, aP1);
    }
    int c = n0 + ln16;
    float bgv = bg[c], bpv = bp[c];
#pragma unroll
    for (int mi = 0; mi < 2; ++mi) {
        v8f& g = mi ? aG1 : aG0;
        v8f& pp = mi ? aP1 : aP0;
#pragma unroll
        for (int e = 0; e < 8; ++e) {
            int r = m0 + mi * 16 + e + 8 * hi;
            float val = sigmoidf_(g[e] + bgv) * (pp[e] + bpv);
            size_t idx = (scatterMode == 1)
                       ? (((size_t)c << 16) + (size_t)r)
                       : (((size_t)c << 16) + ((size_t)(r & 255) << 8) + (size_t)(r >> 8));
            outp[idx] = (__bf16)val;
        }
    }
}

// ---------------------------------------------------------------------------
// 2x2-blocked GEMM core: each wave computes a 32x32 output quad.
// EPI: 0 = bf16 out, 1 = bf16 relu out, 2 = fp32 sigmoid out,
//      3 = residual add (optional gmul, optional transposed store).
// ---------------------------------------------------------------------------
template <int EPI>
__device__ __forceinline__ void gemm32x32(
        const __bf16* __restrict__ A, const __bf16* __restrict__ Wt,
        const float* __restrict__ bias, void* __restrict__ outv,
        const float* __restrict__ gmul, int N, int K, int transStore)
{
    int lane = threadIdx.x & 31, hi = lane >> 4, ln16 = lane & 15;
    int tile = blockIdx.x * 8 + (threadIdx.x >> 5);
    int tn = N >> 5;
    int m0 = (tile / tn) << 5, n0 = (tile % tn) << 5;
    v8f a00 = {}, a01 = {}, a10 = {}, a11 = {};
    for (int k0 = 0; k0 < K; k0 += 32) {
        FragU fa0 = load_frag(A,  m0 + ln16,      K, k0, hi);
        FragU fa1 = load_frag(A,  m0 + 16 + ln16, K, k0, hi);
        FragU fb0 = load_frag(Wt, n0 + ln16,      K, k0, hi);
        FragU fb1 = load_frag(Wt, n0 + 16 + ln16, K, k0, hi);
        a00 = wmma_bf16(fa0, fb0, a00);
        a01 = wmma_bf16(fa0, fb1, a01);
        a10 = wmma_bf16(fa1, fb0, a10);
        a11 = wmma_bf16(fa1, fb1, a11);
    }
    float bv0 = bias[n0 + ln16], bv1 = bias[n0 + 16 + ln16];
#pragma unroll
    for (int mi = 0; mi < 2; ++mi) {
#pragma unroll
        for (int ni = 0; ni < 2; ++ni) {
            v8f& acc = mi ? (ni ? a11 : a10) : (ni ? a01 : a00);
            int c = n0 + ni * 16 + ln16;
            float bv = ni ? bv1 : bv0;
#pragma unroll
            for (int e = 0; e < 8; ++e) {
                int r = m0 + mi * 16 + e + 8 * hi;
                float val = acc[e] + bv;
                if (EPI == 0) {
                    ((__bf16*)outv)[(size_t)r * N + c] = (__bf16)val;
                } else if (EPI == 1) {
                    ((__bf16*)outv)[(size_t)r * N + c] = (__bf16)fmaxf(val, 0.0f);
                } else if (EPI == 2) {
                    ((float*)outv)[(size_t)r * N + c] = sigmoidf_(val);
                } else {
                    if (gmul) val *= gmul[(size_t)r * N + c];
                    size_t idx = transStore
                               ? ((size_t)((r & 255) * 256 + (r >> 8)) * N + c)
                               : ((size_t)r * N + c);
                    ((float*)outv)[idx] += val;
                }
            }
        }
    }
}

__global__ __launch_bounds__(256) void gemm_plain_kernel(
        const __bf16* __restrict__ A, const __bf16* __restrict__ Wt,
        const float* __restrict__ bias, __bf16* __restrict__ out,
        int N, int K, int relu)
{
    if (relu) gemm32x32<1>(A, Wt, bias, out, nullptr, N, K, 0);
    else      gemm32x32<0>(A, Wt, bias, out, nullptr, N, K, 0);
}

__global__ __launch_bounds__(256) void gemm_sigmoid_kernel(
        const __bf16* __restrict__ A, const __bf16* __restrict__ Wt,
        const float* __restrict__ bias, float* __restrict__ out, int N, int K)
{
    gemm32x32<2>(A, Wt, bias, out, nullptr, N, K, 0);
}

__global__ __launch_bounds__(256) void gemm_outres_kernel(
        const __bf16* __restrict__ A, const __bf16* __restrict__ Wt,
        const float* __restrict__ bias, const float* __restrict__ gmul,
        float* __restrict__ resid, int N, int K, int transStore)
{
    gemm32x32<3>(A, Wt, bias, resid, gmul, N, K, transStore);
}

// ---------------------------------------------------------------------------
// Per-channel triangle einsum: X[c][i][j] = sum_k A'[c][i][k] * B'[c][j][k].
// 2x2 blocking; grid = (8, 128): blockIdx.y = channel, 64 32x32 quads/channel.
// ---------------------------------------------------------------------------
__global__ __launch_bounds__(256) void tri_einsum_kernel(
        const __bf16* __restrict__ ab, const __bf16* __restrict__ bb,
        float* __restrict__ xf)
{
    int c = blockIdx.y;
    int lane = threadIdx.x & 31, hi = lane >> 4, ln16 = lane & 15;
    int tile = blockIdx.x * 8 + (threadIdx.x >> 5);
    int m0 = (tile >> 3) << 5, n0 = (tile & 7) << 5;
    const __bf16* A = ab + (size_t)c * LL;
    const __bf16* B = bb + (size_t)c * LL;
    v8f a00 = {}, a01 = {}, a10 = {}, a11 = {};
    for (int k0 = 0; k0 < 256; k0 += 32) {
        FragU fa0 = load_frag(A, m0 + ln16,      256, k0, hi);
        FragU fa1 = load_frag(A, m0 + 16 + ln16, 256, k0, hi);
        FragU fb0 = load_frag(B, n0 + ln16,      256, k0, hi);
        FragU fb1 = load_frag(B, n0 + 16 + ln16, 256, k0, hi);
        a00 = wmma_bf16(fa0, fb0, a00);
        a01 = wmma_bf16(fa0, fb1, a01);
        a10 = wmma_bf16(fa1, fb0, a10);
        a11 = wmma_bf16(fa1, fb1, a11);
    }
    float* xc = xf + (size_t)c * LL;
#pragma unroll
    for (int mi = 0; mi < 2; ++mi) {
#pragma unroll
        for (int ni = 0; ni < 2; ++ni) {
            v8f& acc = mi ? (ni ? a11 : a10) : (ni ? a01 : a00);
#pragma unroll
            for (int e = 0; e < 8; ++e)
                xc[(size_t)(m0 + mi * 16 + e + 8 * hi) * 256 + n0 + ni * 16 + ln16] = acc[e];
        }
    }
}

// bias projection (N=4 too small for WMMA): out[h][x*256+y] = zl[x,y,:]@w[:,h]+b[h]
__global__ __launch_bounds__(256) void biasproj_kernel(const __bf16* __restrict__ zl,
        const float* __restrict__ w, const float* __restrict__ b,
        float* __restrict__ out)
{
    int t = blockIdx.x * 256 + threadIdx.x;   // 65536*4 threads
    int h = t & 3, r = t >> 2;
    float s = b[h];
    for (int c = 0; c < 128; ++c)
        s += (float)zl[(size_t)r * 128 + c] * w[c * 4 + h];
    out[(size_t)h * LL + r] = s;
}

// z[i][j][:] -> zT[j][i][:] (float4 vectorized)
__global__ __launch_bounds__(256) void transpose_kernel(const float* __restrict__ in,
                                                        float* __restrict__ out)
{
    int t = blockIdx.x * 256 + threadIdx.x;   // 2^21 threads
    int c4 = t & 31, j = (t >> 5) & 255, i = t >> 13;
    ((float4*)out)[(((size_t)j * 256 + i) << 5) + c4] =
        ((const float4*)in)[(((size_t)i * 256 + j) << 5) + c4];
}

// ---------------------------------------------------------------------------
// Flash triangle attention. One workgroup = one (head h, row i).
// 8 waves, each owning two 16-row j-tiles; k streamed in 32-wide blocks.
// S = Q K^T via 2 WMMAs, online softmax (16-lane shuffles), P repacked via a
// per-wave LDS tile into A-fragment layout, O += P V via 2 WMMAs (V^T in LDS).
// ---------------------------------------------------------------------------
__global__ __launch_bounds__(256) void att_kernel(
        const __bf16* __restrict__ qb, const __bf16* __restrict__ kb,
        const __bf16* __restrict__ vb, const float* __restrict__ biasb,
        const float* __restrict__ gateb, __bf16* __restrict__ gob)
{
    __shared__ __bf16 Vt[32 * 256];   // V^T: [d][kk], kk contiguous (16 KB)
    __shared__ __bf16 Pt[8][16 * 32]; // per-wave P repack tile (8 KB)

    int h = blockIdx.x & 3;
    int i = blockIdx.x >> 2;
    int wave = threadIdx.x >> 5, lane = threadIdx.x & 31;
    int hi = lane >> 4, ln16 = lane & 15;

    for (int t = threadIdx.x; t < 256 * 32; t += 256) {
        int kk = t >> 5, d = t & 31;
        Vt[d * 256 + kk] = vb[((size_t)(i * 256 + kk)) * 128 + h * 32 + d];
    }
    __syncthreads();

    const __bf16* Qb = qb + (size_t)i * 256 * 128 + h * 32;  // row j, stride 128
    const __bf16* Kb = kb + (size_t)i * 256 * 128 + h * 32;  // row k, stride 128
    const float*  Bh = biasb + (size_t)h * LL;
    const float scale = 0.17677669529663689f;                // 1/sqrt(32)

    for (int jt = wave; jt < 16; jt += 8) {
        int j0 = jt * 16;
        FragU fq = load_frag(Qb, j0 + ln16, 128, 0, hi);     // Q: 16 x 32(d)
        v8f o0 = {}; v8f o1 = {};
        float m[8], l[8];
#pragma unroll
        for (int e = 0; e < 8; ++e) { m[e] = -1e30f; l[e] = 0.0f; }

        for (int k0 = 0; k0 < 256; k0 += 32) {
            FragU fk0 = load_frag(Kb, k0 + ln16, 128, 0, hi);
            FragU fk1 = load_frag(Kb, k0 + 16 + ln16, 128, 0, hi);
            v8f s0 = {}; v8f s1 = {};
            s0 = wmma_bf16(fq, fk0, s0);
            s1 = wmma_bf16(fq, fk1, s1);
#pragma unroll
            for (int e = 0; e < 8; ++e) {
                int j = j0 + e + 8 * hi;
                float sv0 = s0[e] * scale + Bh[j * 256 + k0 + ln16];
                float sv1 = s1[e] * scale + Bh[j * 256 + k0 + 16 + ln16];
                float mx = fmaxf(sv0, sv1);
#pragma unroll
                for (int off = 1; off < 16; off <<= 1)
                    mx = fmaxf(mx, __shfl_xor(mx, off, 32));
                float mn = fmaxf(m[e], mx);
                float corr = __expf(m[e] - mn);
                float p0 = __expf(sv0 - mn);
                float p1 = __expf(sv1 - mn);
                float ps = p0 + p1;
#pragma unroll
                for (int off = 1; off < 16; off <<= 1)
                    ps += __shfl_xor(ps, off, 32);
                l[e] = l[e] * corr + ps;
                m[e] = mn;
                o0[e] *= corr; o1[e] *= corr;
                int row = e + 8 * hi;
                Pt[wave][row * 32 + ln16]      = (__bf16)p0;
                Pt[wave][row * 32 + 16 + ln16] = (__bf16)p1;
            }
            asm volatile("s_wait_dscnt 0" ::: "memory");   // P stores visible
            FragU fp  = load_frag(&Pt[wave][0], ln16, 32, 0, hi);  // P: 16x32(k)
            FragU fv0 = load_frag(Vt, ln16, 256, k0, hi);          // V^T d 0..15
            FragU fv1 = load_frag(Vt, 16 + ln16, 256, k0, hi);     // V^T d 16..31
            o0 = wmma_bf16(fp, fv0, o0);
            o1 = wmma_bf16(fp, fv1, o1);
        }
#pragma unroll
        for (int e = 0; e < 8; ++e) {
            int j = j0 + e + 8 * hi;
            float inv = 1.0f / l[e];
            size_t base = ((size_t)(i * 256 + j)) * 128 + h * 32;
            gob[base + ln16]      = (__bf16)(o0[e] * inv * gateb[base + ln16]);
            gob[base + 16 + ln16] = (__bf16)(o1[e] * inv * gateb[base + 16 + ln16]);
        }
    }
}

// ---------------------------------------------------------------------------
// Host orchestration. Input leaf order (flattened setup_inputs dict):
//  0:z  1..16:tri_out  17..32:tri_in  33..46:att_start  47..60:att_end
//  61..66:trans    (each lin = w,b ; each ln = g,b)
// Workspace budget ~146 MB with reuse.
// ---------------------------------------------------------------------------
extern "C" void kernel_launch(void* const* d_in, const int* in_sizes, int n_in,
                              void* d_out, int out_size, void* d_ws, size_t ws_size,
                              hipStream_t stream)
{
    (void)in_sizes; (void)n_in; (void)out_size; (void)ws_size;
    const float* z_in = (const float*)d_in[0];
    float* zacc = (float*)d_out;
    char* ws = (char*)d_ws;
    const size_t MB = 1024 * 1024;

    __bf16* WT    = (__bf16*)(ws + 0);        //  1 MB  weights bf16 [N][K]
    __bf16* ZL    = (__bf16*)(ws +   1 * MB); // 16 MB  LN(z) bf16
    __bf16* ABUF  = (__bf16*)(ws +  17 * MB); // 16 MB  a' / q / h1-chunk
    __bf16* BBUF  = (__bf16*)(ws +  33 * MB); // 16 MB  b' / k
    __bf16* XL    = (__bf16*)(ws +  49 * MB); // 16 MB  LN(x) / v
    __bf16* GO    = (__bf16*)(ws +  65 * MB); // 16 MB  gate*o (attention)
    float*  GF    = (float*) (ws +  81 * MB); // 32 MB  sigmoid gates fp32
    float*  XF    = (float*) (ws + 113 * MB); // 32 MB  triangle product / zT
    float*  BIASF = (float*) (ws + 145 * MB); //  1 MB  attention bias [h][j][k]

    hipMemcpyAsync(zacc, z_in, (size_t)LL * ZC * sizeof(float),
                   hipMemcpyDeviceToDevice, stream);

    size_t woff = 0;
    auto conv = [&](int idx, int K, int N) -> __bf16* {
        __bf16* dst = WT + woff;
        int total = K * N;
        wconv_kernel<<<(total + 255) / 256, 256, 0, stream>>>(
            (const float*)d_in[idx], dst, K, N);
        woff += (size_t)total;
        return dst;
    };
    __bf16 *to_ap = conv(3,128,128),  *to_ag = conv(5,128,128),
           *to_bp = conv(7,128,128),  *to_bg = conv(9,128,128),
           *to_g  = conv(11,128,128), *to_o  = conv(15,128,128);
    __bf16 *ti_ap = conv(19,128,128), *ti_ag = conv(21,128,128),
           *ti_bp = conv(23,128,128), *ti_bg = conv(25,128,128),
           *ti_g  = conv(27,128,128), *ti_o  = conv(31,128,128);
    __bf16 *as_q = conv(35,128,128), *as_k = conv(37,128,128),
           *as_v = conv(39,128,128), *as_gt = conv(43,128,128),
           *as_o = conv(45,128,128);
    __bf16 *ae_q = conv(49,128,128), *ae_k = conv(51,128,128),
           *ae_v = conv(53,128,128), *ae_gt = conv(57,128,128),
           *ae_o = conv(59,128,128);
    __bf16 *tr_w1 = conv(63,128,512), *tr_w2 = conv(65,512,128);

    auto F = [&](int idx) { return (const float*)d_in[idx]; };

    // grid helper: 32x32 quads, 8 waves per block
    auto blocks32 = [](int M, int N) { return (M / 32) * (N / 32) / 8; };

    // ---- triangle multiplicative update (base=leaf index of ln_in.g) ----
    auto tri_mul = [&](int base, __bf16* wAg, __bf16* wAp, __bf16* wBg,
                       __bf16* wBp, __bf16* wG, __bf16* wO, int mode) {
        ln_kernel<<<8192, 256, 0, stream>>>(zacc, F(base), F(base + 1), ZL, 1);
        gemm_gated_kernel<<<2048, 256, 0, stream>>>(ZL, wAg, F(base + 5),
                                                    wAp, F(base + 3), ABUF, mode);
        gemm_gated_kernel<<<2048, 256, 0, stream>>>(ZL, wBg, F(base + 9),
                                                    wBp, F(base + 7), BBUF, mode);
        gemm_sigmoid_kernel<<<blocks32(LL,128), 256, 0, stream>>>(ZL, wG, F(base + 11),
                                                                  GF, 128, 128);
        tri_einsum_kernel<<<dim3(8, 128), 256, 0, stream>>>(ABUF, BBUF, XF);
        ln_kernel<<<8192, 256, 0, stream>>>(XF, F(base + 12), F(base + 13), XL, LL);
        gemm_outres_kernel<<<blocks32(LL,128), 256, 0, stream>>>(XL, wO, F(base + 15),
                                                                 GF, zacc, 128, 128, 0);
    };
    tri_mul(1,  to_ag, to_ap, to_bg, to_bp, to_g, to_o, 1);  // outgoing
    tri_mul(17, ti_ag, ti_ap, ti_bg, ti_bp, ti_g, ti_o, 2);  // incoming

    // ---- triangle attention (base=leaf index of ln.g) ----
    auto tri_att = [&](int base, __bf16* wq, __bf16* wk, __bf16* wv,
                       __bf16* wg, __bf16* wo, int endNode) {
        const float* zsrc = zacc;
        if (endNode) {
            transpose_kernel<<<8192, 256, 0, stream>>>(zacc, XF);
            zsrc = XF;
        }
        ln_kernel<<<8192, 256, 0, stream>>>(zsrc, F(base), F(base + 1), ZL, 1);
        gemm_plain_kernel<<<blocks32(LL,128), 256, 0, stream>>>(ZL, wq, F(base + 3), ABUF, 128, 128, 0);
        gemm_plain_kernel<<<blocks32(LL,128), 256, 0, stream>>>(ZL, wk, F(base + 5), BBUF, 128, 128, 0);
        gemm_plain_kernel<<<blocks32(LL,128), 256, 0, stream>>>(ZL, wv, F(base + 7), XL,   128, 128, 0);
        gemm_sigmoid_kernel<<<blocks32(LL,128), 256, 0, stream>>>(ZL, wg, F(base + 11), GF, 128, 128);
        biasproj_kernel<<<1024, 256, 0, stream>>>(ZL, F(base + 8), F(base + 9), BIASF);
        att_kernel<<<1024, 256, 0, stream>>>(ABUF, BBUF, XL, BIASF, GF, GO);
        gemm_outres_kernel<<<blocks32(LL,128), 256, 0, stream>>>(GO, wo, F(base + 13),
                                                                 nullptr, zacc, 128, 128, endNode);
    };
    tri_att(33, as_q, as_k, as_v, as_gt, as_o, 0);  // starting node
    tri_att(47, ae_q, ae_k, ae_v, ae_gt, ae_o, 1);  // ending node (on z^T)

    // ---- transition (chunked over rows so h1 fits the 16MB scratch tile) ----
    ln_kernel<<<8192, 256, 0, stream>>>(zacc, F(61), F(62), ZL, 1);
    for (int chunk = 0; chunk < 4; ++chunk) {
        const __bf16* Ach = ZL + (size_t)chunk * 16384 * 128;
        gemm_plain_kernel<<<blocks32(16384,512), 256, 0, stream>>>(Ach, tr_w1, F(64),
                                                                   ABUF, 512, 128, 1);   // relu
        gemm_outres_kernel<<<blocks32(16384,128), 256, 0, stream>>>(ABUF, tr_w2, F(66), nullptr,
                                                                    zacc + (size_t)chunk * 16384 * 128,
                                                                    128, 512, 0);
    }
}
